// JASSAttentionLayer_24988119728571
// MI455X (gfx1250) — matrix operand order repaired and checked by reading, not compiled, not run on hardware
//
#include <hip/hip_runtime.h>
#include <hip/hip_bf16.h>

typedef _Float16 h4  __attribute__((ext_vector_type(4)));
typedef _Float16 h8  __attribute__((ext_vector_type(8)));
typedef _Float16 h16 __attribute__((ext_vector_type(16)));
typedef float    f4v __attribute__((ext_vector_type(4)));
typedef float    f8  __attribute__((ext_vector_type(8)));
typedef int      i4  __attribute__((ext_vector_type(4)));

// address-space-qualified int4 for the async-to-LDS builtin
typedef __attribute__((address_space(1))) i4 gi4;   // global
typedef __attribute__((address_space(3))) i4 li4;   // LDS

#define HDIM  1024
#define BB    4
#define SS    1024
#define NHEAD 16
#define DHEAD 64
#define MROWS (BB * SS)
#define LDSA_STRIDE 40   // 32 halfs + 8 pad
#define LDSK_STRIDE 72   // 64 halfs + 8 pad

#if defined(__AMDGCN__) && __has_builtin(__builtin_amdgcn_global_load_async_to_lds_b128)
#define HAVE_ASYNC_LDS 1
#else
#define HAVE_ASYNC_LDS 0
#endif

__device__ __forceinline__ h16 make_h16(h8 lo, h8 hi) {
  return __builtin_shufflevector(lo, hi, 0,1,2,3,4,5,6,7,8,9,10,11,12,13,14,15);
}

__device__ __forceinline__ void wait_async_lds() {
#if HAVE_ASYNC_LDS
#if __has_builtin(__builtin_amdgcn_s_wait_asynccnt)
  __builtin_amdgcn_s_wait_asynccnt(0);
#else
  asm volatile("s_wait_asynccnt 0x0" ::: "memory");
#endif
#endif
}

// ---------------------------------------------------------------------------
// Projection GEMM: out = concat_l(A_l) @ concat_l(W_l), f32 in -> f16 out.
// MODE 0: out[m][n] f16 row-major [MROWS][HDIM] (scaled)
// MODE 1: transposed-per-head store -> out[((b*NH+h)*DHEAD+dd)][key]
// Block tile 128x128, wave tile 32x64, K step 32, double-buffered LDS.
// ---------------------------------------------------------------------------
template <int MODE>
__global__ __launch_bounds__(256) void proj_gemm(
    const float* __restrict__ A, const float* __restrict__ W,
    _Float16* __restrict__ out, int links, float scale)
{
  __shared__ _Float16 lds_a[2][128 * LDSA_STRIDE];
  __shared__ _Float16 lds_b[2][128 * LDSA_STRIDE];

  const int tid  = threadIdx.x;
  const int wave = tid >> 5, lane = tid & 31;
  const int lm = lane & 15, g = lane >> 4;
  const int bm = blockIdx.x * 128;
  const int bn = blockIdx.y * 128;
  const int waveM = (wave & 3) * 32;
  const int waveN = (wave >> 2) * 64;

  auto load_tile = [&](int kt, int buf) {
    const int link = (kt * 32) / HDIM;
    const int colk = (kt * 32) % HDIM;
    const float* Ab = A + (size_t)link * MROWS * HDIM;
    const float* Wb = W + (size_t)link * HDIM * HDIM;
    // A tile: 128 rows x 32 cols f32 -> f16, row-major LDS
#pragma unroll
    for (int i = 0; i < 4; ++i) {
      int f = tid + i * 256;            // 0..1023 float4s
      int row = f >> 3, c4 = f & 7;
      f4v v = *(const f4v*)(Ab + (size_t)(bm + row) * HDIM + colk + c4 * 4);
      h4 hv;
      hv[0] = (_Float16)v[0]; hv[1] = (_Float16)v[1];
      hv[2] = (_Float16)v[2]; hv[3] = (_Float16)v[3];
      *(h4*)&lds_a[buf][row * LDSA_STRIDE + c4 * 4] = hv;
    }
    // B tile: 32 k x 128 n. Each thread: 4x4 (k x n) block, register
    // transpose, 4 packed b64 stores into transposed [n][k] LDS tile.
    {
      int n4  = (tid & 31) * 4;
      int kk0 = (tid >> 5) * 4;
      f4v rows[4];
#pragma unroll
      for (int j = 0; j < 4; ++j)
        rows[j] = *(const f4v*)(Wb + (size_t)(colk + kk0 + j) * HDIM + bn + n4);
#pragma unroll
      for (int jn = 0; jn < 4; ++jn) {
        h4 hv;
#pragma unroll
        for (int jk = 0; jk < 4; ++jk) hv[jk] = (_Float16)rows[jk][jn];
        *(h4*)&lds_b[buf][(n4 + jn) * LDSA_STRIDE + kk0] = hv;
      }
    }
  };

  const f8 fz = {};
  f8 acc[2][4];
#pragma unroll
  for (int i = 0; i < 2; ++i)
#pragma unroll
    for (int j = 0; j < 4; ++j) acc[i][j] = fz;

  const int ktiles = links * (HDIM / 32);
  load_tile(0, 0);
  __syncthreads();

  for (int kt = 0; kt < ktiles; ++kt) {
    const int cur = kt & 1;
    if (kt + 1 < ktiles) load_tile(kt + 1, cur ^ 1);

    h16 af[2], bf[4];
#pragma unroll
    for (int ms = 0; ms < 2; ++ms) {
      int row = waveM + ms * 16 + lm;
      h8 lo = *(const h8*)&lds_a[cur][row * LDSA_STRIDE + 8 * g];
      h8 hi = *(const h8*)&lds_a[cur][row * LDSA_STRIDE + 16 + 8 * g];
      af[ms] = make_h16(lo, hi);
    }
#pragma unroll
    for (int ns = 0; ns < 4; ++ns) {
      int n = waveN + ns * 16 + lm;
      h8 lo = *(const h8*)&lds_b[cur][n * LDSA_STRIDE + 8 * g];
      h8 hi = *(const h8*)&lds_b[cur][n * LDSA_STRIDE + 16 + 8 * g];
      bf[ns] = make_h16(lo, hi);
    }
#pragma unroll
    for (int ms = 0; ms < 2; ++ms)
#pragma unroll
      for (int ns = 0; ns < 4; ++ns)
        acc[ms][ns] = __builtin_amdgcn_wmma_f32_16x16x32_f16(
            false, af[ms], false, bf[ns], (short)0, acc[ms][ns], false, false);
    __syncthreads();
  }

#pragma unroll
  for (int ms = 0; ms < 2; ++ms)
#pragma unroll
    for (int ns = 0; ns < 4; ++ns)
#pragma unroll
      for (int r = 0; r < 8; ++r) {
        int gm = bm + waveM + ms * 16 + r + 8 * g;
        int gn = bn + waveN + ns * 16 + lm;
        _Float16 v = (_Float16)(acc[ms][ns][r] * scale);
        if (MODE == 0) {
          out[(size_t)gm * HDIM + gn] = v;
        } else {
          int b = gm >> 10, key = gm & 1023;
          int h = gn >> 6,  dd  = gn & 63;
          out[((size_t)((b * NHEAD + h) * DHEAD + dd)) * SS + key] = v;
        }
      }
}

// ---------------------------------------------------------------------------
// Output GEMM: d_out = ctx(f16 [MROWS][HDIM]) @ Wo(f32) -> f32
// ---------------------------------------------------------------------------
__global__ __launch_bounds__(256) void out_gemm(
    const _Float16* __restrict__ A, const float* __restrict__ W,
    float* __restrict__ out)
{
  __shared__ _Float16 lds_a[2][128 * LDSA_STRIDE];
  __shared__ _Float16 lds_b[2][128 * LDSA_STRIDE];

  const int tid  = threadIdx.x;
  const int wave = tid >> 5, lane = tid & 31;
  const int lm = lane & 15, g = lane >> 4;
  const int bm = blockIdx.x * 128;
  const int bn = blockIdx.y * 128;
  const int waveM = (wave & 3) * 32;
  const int waveN = (wave >> 2) * 64;

  auto load_tile = [&](int kt, int buf) {
    const int colk = kt * 32;
#pragma unroll
    for (int i = 0; i < 2; ++i) {
      int f = tid + i * 256;            // 0..511 h8 chunks
      int row = f >> 2, c8 = f & 3;
      h8 v = *(const h8*)(A + (size_t)(bm + row) * HDIM + colk + c8 * 8);
      *(h8*)&lds_a[buf][row * LDSA_STRIDE + c8 * 8] = v;
    }
    {
      int n4  = (tid & 31) * 4;
      int kk0 = (tid >> 5) * 4;
      f4v rows[4];
#pragma unroll
      for (int j = 0; j < 4; ++j)
        rows[j] = *(const f4v*)(W + (size_t)(colk + kk0 + j) * HDIM + bn + n4);
#pragma unroll
      for (int jn = 0; jn < 4; ++jn) {
        h4 hv;
#pragma unroll
        for (int jk = 0; jk < 4; ++jk) hv[jk] = (_Float16)rows[jk][jn];
        *(h4*)&lds_b[buf][(n4 + jn) * LDSA_STRIDE + kk0] = hv;
      }
    }
  };

  const f8 fz = {};
  f8 acc[2][4];
#pragma unroll
  for (int i = 0; i < 2; ++i)
#pragma unroll
    for (int j = 0; j < 4; ++j) acc[i][j] = fz;

  load_tile(0, 0);
  __syncthreads();

  for (int kt = 0; kt < HDIM / 32; ++kt) {
    const int cur = kt & 1;
    if (kt + 1 < HDIM / 32) load_tile(kt + 1, cur ^ 1);

    h16 af[2], bf[4];
#pragma unroll
    for (int ms = 0; ms < 2; ++ms) {
      int row = waveM + ms * 16 + lm;
      h8 lo = *(const h8*)&lds_a[cur][row * LDSA_STRIDE + 8 * g];
      h8 hi = *(const h8*)&lds_a[cur][row * LDSA_STRIDE + 16 + 8 * g];
      af[ms] = make_h16(lo, hi);
    }
#pragma unroll
    for (int ns = 0; ns < 4; ++ns) {
      int n = waveN + ns * 16 + lm;
      h8 lo = *(const h8*)&lds_b[cur][n * LDSA_STRIDE + 8 * g];
      h8 hi = *(const h8*)&lds_b[cur][n * LDSA_STRIDE + 16 + 8 * g];
      bf[ns] = make_h16(lo, hi);
    }
#pragma unroll
    for (int ms = 0; ms < 2; ++ms)
#pragma unroll
      for (int ns = 0; ns < 4; ++ns)
        acc[ms][ns] = __builtin_amdgcn_wmma_f32_16x16x32_f16(
            false, af[ms], false, bf[ns], (short)0, acc[ms][ns], false, false);
    __syncthreads();
  }

#pragma unroll
  for (int ms = 0; ms < 2; ++ms)
#pragma unroll
    for (int ns = 0; ns < 4; ++ns)
#pragma unroll
      for (int r = 0; r < 8; ++r) {
        int gm = bm + waveM + ms * 16 + r + 8 * g;
        int gn = bn + waveN + ns * 16 + lm;
        out[(size_t)gm * HDIM + gn] = acc[ms][ns][r];
      }
}

// ---------------------------------------------------------------------------
// Flash attention per (b, head). 8 waves x 32 query rows (2 sub-tiles of 16)
// = 256 q rows per block. K/V tiles staged in double-buffered LDS via
// GLOBAL_LOAD_ASYNC_TO_LDS (ASYNCcnt) when available; shared by all waves.
// S^T = K.q^T (softmax rows in a lane pair), online softmax, O^T = V^T.P^T.
// ---------------------------------------------------------------------------
__global__ __launch_bounds__(256) void attn_kernel(
    const _Float16* __restrict__ q, const _Float16* __restrict__ k,
    const _Float16* __restrict__ vT, const float* __restrict__ bias,
    _Float16* __restrict__ ctx, float lmul)
{
  __shared__ _Float16 lds_k[2][32 * LDSK_STRIDE];   // 32 keys x 64 d
  __shared__ _Float16 lds_v[2][64 * LDSA_STRIDE];   // 64 d x 32 keys (V^T)

  const int tid  = threadIdx.x;
  const int wave = tid >> 5, lane = tid & 31;
  const int lm = lane & 15, g = lane >> 4;
  const int qblk = blockIdx.x & 3;
  const int bh   = blockIdx.x >> 2;
  const int b = bh >> 4, h = bh & 15;

  const size_t qrow0 = (size_t)(b * SS + qblk * 256 + wave * 32);
  const _Float16* qp = q  + qrow0 * HDIM + h * DHEAD;
  const _Float16* kp = k  + (size_t)(b * SS) * HDIM + h * DHEAD;
  const _Float16* vp = vT + (size_t)(b * NHEAD + h) * DHEAD * SS;
  const float*    bp = bias + b * SS;
  _Float16*       cp = ctx + qrow0 * HDIM + h * DHEAD;

  const int krow = tid >> 3, kc8 = tid & 7;   // K tile: 32 rows x 64 halfs
  const int vrow = tid >> 2, vc8 = tid & 3;   // V^T tile: 64 rows x 32 halfs

  auto load_kv = [&](int kb, int buf) {
#if HAVE_ASYNC_LDS
    __builtin_amdgcn_global_load_async_to_lds_b128(
        (gi4*)(kp + (size_t)(kb + krow) * HDIM + kc8 * 8),
        (li4*)&lds_k[buf][krow * LDSK_STRIDE + kc8 * 8],
        0, 0);
    __builtin_amdgcn_global_load_async_to_lds_b128(
        (gi4*)(vp + (size_t)vrow * SS + kb + vc8 * 8),
        (li4*)&lds_v[buf][vrow * LDSA_STRIDE + vc8 * 8],
        0, 0);
#else
    h8 kv = *(const h8*)(kp + (size_t)(kb + krow) * HDIM + kc8 * 8);
    *(h8*)&lds_k[buf][krow * LDSK_STRIDE + kc8 * 8] = kv;
    h8 vv = *(const h8*)(vp + (size_t)vrow * SS + kb + vc8 * 8);
    *(h8*)&lds_v[buf][vrow * LDSA_STRIDE + vc8 * 8] = vv;
#endif
  };

  // q^T B-fragments, 2 q sub-tiles (lane lm = query row within sub-tile)
  h16 qb[2][2];
#pragma unroll
  for (int u = 0; u < 2; ++u)
#pragma unroll
    for (int s = 0; s < 2; ++s) {
      const _Float16* base = qp + (size_t)(u * 16 + lm) * HDIM + s * 32 + 8 * g;
      qb[u][s] = make_h16(*(const h8*)base, *(const h8*)(base + 16));
    }

  const f8 fz = {};
  f8 acc[2][4];
#pragma unroll
  for (int u = 0; u < 2; ++u)
#pragma unroll
    for (int d = 0; d < 4; ++d) acc[u][d] = fz;
  float mrun[2] = {-1e30f, -1e30f}, srun[2] = {0.0f, 0.0f};

  load_kv(0, 0);
  wait_async_lds();
  __syncthreads();

  for (int kb = 0; kb < SS; kb += 32) {
    const int cur = (kb >> 5) & 1;
    if (kb + 32 < SS) load_kv(kb + 32, cur ^ 1);

    // S^T tiles: st[u][t]: keys kb+16t..+15 (M) x 16 queries of sub-tile u
    f8 st[2][2];
#pragma unroll
    for (int u = 0; u < 2; ++u) { st[u][0] = fz; st[u][1] = fz; }
#pragma unroll
    for (int t = 0; t < 2; ++t)
#pragma unroll
      for (int s = 0; s < 2; ++s) {
        const _Float16* base =
            &lds_k[cur][(t * 16 + lm) * LDSK_STRIDE + s * 32 + 8 * g];
        h16 ka = make_h16(*(const h8*)base, *(const h8*)(base + 16));
#pragma unroll
        for (int u = 0; u < 2; ++u)
          st[u][t] = __builtin_amdgcn_wmma_f32_16x16x32_f16(
              false, ka, false, qb[u][s], (short)0, st[u][t], false, false);
      }
    // + L * bias[key]; key of element r = kb + 16t + 8g + r
#pragma unroll
    for (int t = 0; t < 2; ++t) {
      const float* bb = bp + kb + t * 16 + 8 * g;
      f4v b0 = *(const f4v*)bb;
      f4v b1 = *(const f4v*)(bb + 4);
#pragma unroll
      for (int u = 0; u < 2; ++u)
#pragma unroll
        for (int r = 0; r < 4; ++r) {
          st[u][t][r]     += lmul * b0[r];
          st[u][t][r + 4] += lmul * b1[r];
        }
    }

    h16 pb[2];
#pragma unroll
    for (int u = 0; u < 2; ++u) {
      // online softmax over keys (per query = lane pair {lane, lane^16})
      float mb = -1e30f;
#pragma unroll
      for (int r = 0; r < 8; ++r)
        mb = fmaxf(mb, fmaxf(st[u][0][r], st[u][1][r]));
      mb = fmaxf(mb, __shfl_xor(mb, 16, 32));
      float mnew = fmaxf(mrun[u], mb);
      float corr = __expf(mrun[u] - mnew);
      float p0[8], p1[8], ls = 0.0f;
#pragma unroll
      for (int r = 0; r < 8; ++r) {
        p0[r] = __expf(st[u][0][r] - mnew);
        p1[r] = __expf(st[u][1][r] - mnew);
        ls += p0[r] + p1[r];
      }
      ls += __shfl_xor(ls, 16, 32);
      srun[u] = srun[u] * corr + ls;
      mrun[u] = mnew;
#pragma unroll
      for (int d = 0; d < 4; ++d)
#pragma unroll
        for (int r = 0; r < 8; ++r) acc[u][d][r] *= corr;
      // Rebuild P^T B-fragment (32 keys x 16 q): element j <-> key 16g+j
      h8 blo, bhi;
#pragma unroll
      for (int j = 0; j < 8; ++j) {
        float o0 = __shfl_xor(p0[j], 16, 32);
        float o1 = __shfl_xor(p1[j], 16, 32);
        blo[j] = (_Float16)(g ? o1 : p0[j]);
        bhi[j] = (_Float16)(g ? p1[j] : o0);
      }
      pb[u] = make_h16(blo, bhi);
    }

    // O^T += V^T . P^T  (4 d-blocks of 16; V fragment reused by both u)
#pragma unroll
    for (int d = 0; d < 4; ++d) {
      const _Float16* base = &lds_v[cur][(d * 16 + lm) * LDSA_STRIDE + 16 * g];
      h16 va = make_h16(*(const h8*)base, *(const h8*)(base + 8));
#pragma unroll
      for (int u = 0; u < 2; ++u)
        acc[u][d] = __builtin_amdgcn_wmma_f32_16x16x32_f16(
            false, va, false, pb[u], (short)0, acc[u][d], false, false);
    }
    wait_async_lds();
    __syncthreads();
  }

#pragma unroll
  for (int u = 0; u < 2; ++u) {
    float inv = 1.0f / srun[u];
#pragma unroll
    for (int d = 0; d < 4; ++d) {
      h8 o;
#pragma unroll
      for (int r = 0; r < 8; ++r) o[r] = (_Float16)(acc[u][d][r] * inv);
      *(h8*)(cp + (size_t)(u * 16 + lm) * HDIM + d * 16 + 8 * g) = o;
    }
  }
}

// ---------------------------------------------------------------------------
extern "C" void kernel_launch(void* const* d_in, const int* in_sizes, int n_in,
                              void* d_out, int out_size, void* d_ws, size_t ws_size,
                              hipStream_t stream) {
  const float* x    = (const float*)d_in[0];
  const float* y    = (const float*)d_in[1];
  const float* bias = (const float*)d_in[2];
  const float* Wq   = (const float*)d_in[3];
  const float* Wk   = (const float*)d_in[4];
  const float* Wv   = (const float*)d_in[5];
  const float* Wo   = (const float*)d_in[6];

  const int L = in_sizes[1] / in_sizes[0];  // y is [L,B,S,H], x is [B,S,H]

  _Float16* wsq = (_Float16*)d_ws;
  _Float16* wsk = wsq + (size_t)MROWS * HDIM;
  _Float16* wsv = wsk + (size_t)MROWS * HDIM;
  _Float16* wsc = wsv + (size_t)MROWS * HDIM;

  dim3 grid(MROWS / 128, HDIM / 128), blk(256);
  const float qscale = 0.125f;  // d^-0.5, d = 64

  proj_gemm<0><<<grid, blk, 0, stream>>>(x, Wq, wsq, 1, qscale);
  proj_gemm<0><<<grid, blk, 0, stream>>>(y, Wk, wsk, L, 1.0f);
  proj_gemm<1><<<grid, blk, 0, stream>>>(y, Wv, wsv, L, 1.0f);
  attn_kernel<<<dim3(BB * NHEAD * (SS / 256)), blk, 0, stream>>>(
      wsq, wsk, wsv, bias, wsc, (float)L);
  out_gemm<<<grid, blk, 0, stream>>>(wsc, Wo, (float*)d_out);
}
